// GlobalAttention_65266323030523
// MI455X (gfx1250) — compile-verified, matrix operand order checked by reading, not run on hardware
//
#include <hip/hip_runtime.h>
#include <hip/hip_bf16.h>
#include <math.h>

typedef __attribute__((ext_vector_type(8)))  float  v8f;
typedef __attribute__((ext_vector_type(4)))  __bf16 v4bf;
typedef __attribute__((ext_vector_type(8)))  __bf16 v8bf;
typedef __attribute__((ext_vector_type(16))) __bf16 v16bf;

#define B_   32
#define L_   2048
#define D_   512      // d_global == d_seq == 512
#define H_   8
#define DK_  64
#define NSPLIT 4
#define CHUNK (L_ / NSPLIT)          // 512 rows per split
#define TILES (CHUNK / 16)           // 32 tiles of 16 rows
#define KSTEPS (D_ / 32)             // 16 WMMA K-steps
#define FRAGN (KSTEPS * 4 * 32 * 16) // 32768 bf16 per weight matrix
#define SROW 520                     // padded bf16 row stride (bank-conflict free)

static __device__ inline v16bf join16(v8bf lo, v8bf hi) {
  return __builtin_shufflevector(lo, hi, 0,1,2,3,4,5,6,7,8,9,10,11,12,13,14,15);
}

static __device__ inline float fast_tanh(float x) {
  x = fminf(fmaxf(x, -15.f), 15.f);
  float e = __expf(2.0f * x);
  return (e - 1.0f) / (e + 1.0f);
}

// ---------------------------------------------------------------------------
// Fused: Q = tanh(X Wq); per 16-row tile: Kt=tanh(S Wk), V=relu(S Wv) via
// bf16 WMMA; online softmax over scores = Kt·q / 8; partial (m,s,acc) to ws.
// grid: B*H*NSPLIT blocks x 256 threads (8 waves: 0-3 -> K gemm, 4-7 -> V gemm)
// ---------------------------------------------------------------------------
__global__ void __launch_bounds__(256)
ga_flash_kernel(const float* __restrict__ X, const float* __restrict__ S,
                const float* __restrict__ Wq, const float* __restrict__ Wk,
                const float* __restrict__ Wv,
                float* __restrict__ ws_m, float* __restrict__ ws_s,
                float* __restrict__ ws_acc)
{
  __shared__ __bf16 wfrag[2][FRAGN];     // [0]=Wk frags, [1]=Wv frags (128 KB)
  __shared__ __bf16 s_tile[16 * SROW];   // 16x512(+pad) bf16 A tile (~16.3 KB)
  __shared__ float  sh_x[D_];            // X[b] (2 KB)
  __shared__ float  sh_q[DK_];           // q vector
  __shared__ float  sh_part[4][16];      // per-K-wave score partials
  __shared__ float  sh_p[16];            // softmax weights for the tile
  __shared__ float  sh_scale;            // rescale factor for running acc

  const int tid  = threadIdx.x;
  const int lane = tid & 31;
  const int wave = tid >> 5;
  const int wn   = wave & 3;             // N-tile index (0..3)
  const int sel  = wave >> 2;            // 0 = K-gemm waves, 1 = V-gemm waves

  const int pid = blockIdx.x;            // b*32 + h*4 + sp
  const int sp  = pid & (NSPLIT - 1);
  const int h   = (pid >> 2) & (H_ - 1);
  const int b   = pid >> 5;
  const int l0c = sp * CHUNK;

  // ---- load X[b] to LDS; build swizzled bf16 B-fragments of Wk[h], Wv[h] ----
  for (int i = tid; i < D_; i += 256) sh_x[i] = X[b * D_ + i];

  for (int idx = tid; idx < 2 * FRAGN; idx += 256) {
    int s2 = idx >> 15;                  // which matrix
    int r  = idx & (FRAGN - 1);
    int e  = r & 15;
    int l  = (r >> 4) & 31;
    int w2 = (r >> 9) & 3;
    int kk = r >> 11;
    int koff = ((l < 16) ? 0 : 8) + (e & 7) + ((e >> 3) << 4);
    int K = kk * 32 + koff;
    int n = w2 * 16 + (l & 15);
    const float* W = s2 ? Wv : Wk;
    wfrag[s2][r] = (__bf16)W[(h * D_ + K) * DK_ + n];
  }
  __syncthreads();

  // ---- q = tanh(X[b] · Wq[h]) ----
  if (tid < DK_) {
    float a = 0.f;
    for (int g = 0; g < D_; ++g) a += sh_x[g] * Wq[(h * D_ + g) * DK_ + tid];
    sh_q[tid] = fast_tanh(a);
  }

  // ---- online softmax state ----
  float m_run = -INFINITY;               // valid in threads 0..15
  float s_run = 0.f;
  float acc   = 0.f;                     // V-waves: running output column

  const int m_row = lane & 15;
  const int kbase = (lane < 16) ? 0 : 8;
  const int half  = lane >> 4;

  const size_t sbase = ((size_t)b * L_ + l0c) * D_;

  // register double-buffer of the 16x512 f32 tile (32 floats/thread)
  float4 stage[8];
  #pragma unroll
  for (int j = 0; j < 8; ++j)
    stage[j] = *reinterpret_cast<const float4*>(&S[sbase + tid * 4 + j * 1024]);

  for (int t = 0; t < TILES; ++t) {
    __syncthreads();                     // prev tile fully consumed
    // convert staged f32 -> packed bf16 LDS (8 x ds_store_b64 per thread)
    #pragma unroll
    for (int j = 0; j < 8; ++j) {
      int e0  = tid * 4 + j * 1024;
      int row = e0 >> 9, col = e0 & 511;
      float4 f = stage[j];
      v4bf p4 = { (__bf16)f.x, (__bf16)f.y, (__bf16)f.z, (__bf16)f.w };
      *reinterpret_cast<v4bf*>(&s_tile[row * SROW + col]) = p4;
    }
    // issue next tile's global loads now; latency hides behind the GEMM
    if (t + 1 < TILES) {
      const size_t sb1 = sbase + (size_t)(t + 1) * 16 * D_;
      #pragma unroll
      for (int j = 0; j < 8; ++j)
        stage[j] = *reinterpret_cast<const float4*>(&S[sb1 + tid * 4 + j * 1024]);
    }
    if (t + 2 < TILES) {
      const size_t sb2 = sbase + (size_t)(t + 2) * 16 * D_;
      #pragma unroll
      for (int j = 0; j < 8; ++j)
        __builtin_prefetch(&S[sb2 + tid * 4 + j * 1024], 0, 1);
    }
    __syncthreads();

    // ---- 16x64x512 GEMM (per wave: one 16x16 N-tile, 16 K-steps) ----
    v8f c = {};
    const __bf16* fb_base = &wfrag[sel][0];
    #pragma unroll
    for (int kk = 0; kk < KSTEPS; ++kk) {
      const __bf16* pa = &s_tile[m_row * SROW + kk * 32 + kbase];
      v16bf a = join16(*reinterpret_cast<const v8bf*>(pa),
                       *reinterpret_cast<const v8bf*>(pa + 16));
      const __bf16* pb = fb_base + (((kk * 4 + wn) * 32) + lane) * 16;
      v16bf bf = join16(*reinterpret_cast<const v8bf*>(pb),
                        *reinterpret_cast<const v8bf*>(pb + 8));
      c = __builtin_amdgcn_wmma_f32_16x16x32_bf16(
            false, a, false, bf, (short)0, c, false, false);
    }

    if (sel == 0) {
      // ---- Kt-waves: tanh, dot with q, reduce -> score partials ----
      float qv = sh_q[wn * 16 + m_row];
      float part[8];
      #pragma unroll
      for (int r = 0; r < 8; ++r) part[r] = fast_tanh(c[r]) * qv;
      #pragma unroll
      for (int off = 1; off < 16; off <<= 1) {
        #pragma unroll
        for (int r = 0; r < 8; ++r) part[r] += __shfl_xor(part[r], off);
      }
      if ((lane & 15) == 0) {
        #pragma unroll
        for (int r = 0; r < 8; ++r) sh_part[wn][r + 8 * half] = part[r];
      }
    }
    __syncthreads();

    // ---- online softmax update (threads 0..15, rows of this tile) ----
    if (tid < 16) {
      float sc = (sh_part[0][tid] + sh_part[1][tid] +
                  sh_part[2][tid] + sh_part[3][tid]) * 0.125f; // 1/sqrt(64)
      float mx = sc;
      #pragma unroll
      for (int off = 8; off >= 1; off >>= 1) mx = fmaxf(mx, __shfl_xor(mx, off));
      float m_new = fmaxf(m_run, mx);
      float scale = __expf(m_run - m_new);
      float p     = __expf(sc - m_new);
      float ps = p;
      #pragma unroll
      for (int off = 8; off >= 1; off >>= 1) ps += __shfl_xor(ps, off);
      s_run = s_run * scale + ps;
      m_run = m_new;
      sh_p[tid] = p;
      if (tid == 0) sh_scale = scale;
    }
    __syncthreads();

    // ---- V-waves: acc = acc*scale + sum_rows p[row] * relu(V) ----
    if (sel == 1) {
      float contrib = 0.f;
      #pragma unroll
      for (int r = 0; r < 8; ++r) {
        float v = c[r];
        v = v > 0.f ? v : 0.f;
        contrib += sh_p[r + 8 * half] * v;
      }
      contrib += __shfl_xor(contrib, 16);
      acc = acc * sh_scale + contrib;
    }
  }

  // ---- write split partials ----
  if (sel == 1 && (lane & 16) == 0)
    ws_acc[(size_t)pid * DK_ + wn * 16 + m_row] = acc;
  if (tid == 0) { ws_m[pid] = m_run; ws_s[pid] = s_run; }
}

// ---------------------------------------------------------------------------
// Combine NSPLIT partials per (b,h): Y = sum_i e^{m_i-m*} acc_i / sum_i e^{m_i-m*} s_i
// grid: B*H blocks x 64 threads
// ---------------------------------------------------------------------------
__global__ void __launch_bounds__(64)
ga_combine_kernel(const float* __restrict__ ws_m, const float* __restrict__ ws_s,
                  const float* __restrict__ ws_acc, float* __restrict__ out)
{
  const int bh = blockIdx.x;
  const int k  = threadIdx.x;
  float m = -INFINITY;
  #pragma unroll
  for (int sp = 0; sp < NSPLIT; ++sp) m = fmaxf(m, ws_m[bh * NSPLIT + sp]);
  float num = 0.f, den = 0.f;
  #pragma unroll
  for (int sp = 0; sp < NSPLIT; ++sp) {
    int pid = bh * NSPLIT + sp;
    float w = __expf(ws_m[pid] - m);
    num += w * ws_acc[(size_t)pid * DK_ + k];
    den += w * ws_s[pid];
  }
  out[bh * DK_ + k] = num / den;
}

extern "C" void kernel_launch(void* const* d_in, const int* in_sizes, int n_in,
                              void* d_out, int out_size, void* d_ws, size_t ws_size,
                              hipStream_t stream) {
  (void)in_sizes; (void)n_in; (void)out_size; (void)ws_size;
  const float* X  = (const float*)d_in[0];
  const float* S  = (const float*)d_in[1];
  const float* Wq = (const float*)d_in[2];
  const float* Wk = (const float*)d_in[3];
  const float* Wv = (const float*)d_in[4];

  const int NP = B_ * H_ * NSPLIT;       // 1024 partials
  float* wsf    = (float*)d_ws;
  float* ws_m   = wsf;                   // [NP]
  float* ws_s   = wsf + NP;              // [NP]
  float* ws_acc = wsf + 2 * NP;          // [NP * 64]

  ga_flash_kernel<<<NP, 256, 0, stream>>>(X, S, Wq, Wk, Wv, ws_m, ws_s, ws_acc);
  ga_combine_kernel<<<B_ * H_, 64, 0, stream>>>(ws_m, ws_s, ws_acc, (float*)d_out);
}
